// ReformerLayer_858993459688
// MI455X (gfx1250) — matrix operand order repaired and checked by reading, not compiled, not run on hardware
//
#include <hip/hip_runtime.h>
#include <hip/hip_bf16.h>
#include <math.h>

typedef __attribute__((ext_vector_type(16))) __bf16 v16bf;
typedef __attribute__((ext_vector_type(8)))  float  v8f;

#define Tt 4096
#define Cc 512
#define DH 64
#define BH 32      // B*HEADS
#define BT 16384   // B*T
#define NH 4       // n_hashes
#define NB 64      // n_buckets
#define NCHUNK 256 // NH * T / 64
#define FLATN 16384 // NH * T

static __device__ __forceinline__ unsigned short f2bf_u(float f) {
  union { float f; unsigned u; } x; x.f = f;
  unsigned r = x.u + 0x7fffu + ((x.u >> 16) & 1u);
  return (unsigned short)(r >> 16);
}
static __device__ __forceinline__ float bf2f_u(unsigned short h) {
  union { unsigned u; float f; } x; x.u = ((unsigned)h) << 16;
  return x.f;
}
static __device__ __forceinline__ v8f v8f_zero() {
  v8f z;
#pragma unroll
  for (int i = 0; i < 8; ++i) z[i] = 0.f;
  return z;
}
static __device__ __forceinline__ v8f wmma_bf16(v16bf a, v16bf b, v8f c) {
  return __builtin_amdgcn_wmma_f32_16x16x32_bf16(false, a, false, b, (short)0, c, false, false);
}

// A fragment 16x32 bf16; A[m][k] = base[(mbase+m)*stride + kbase+k].
// Order used here (contiguous, same as B): M = lane&15; K = (lane>=16?16:0) + 2j + h.
// Per lane the 16 values are two contiguous 16-byte runs -> 2x ds_load_b128.
static __device__ __forceinline__ v16bf load_a_frag(const unsigned short* base, int stride,
                                                    int mbase, int kbase, int lane) {
  int m = mbase + (lane & 15);
  int ko = (lane & 16) ? 16 : 0;
  const unsigned short* p = &base[m * stride + kbase + ko];
  uint4 t[2];
  t[0] = *(const uint4*)p;          // K = kbase+ko   .. +7   (j=0..3)
  t[1] = *(const uint4*)(p + 8);    // K = kbase+ko+8 .. +15  (j=4..7)
  v16bf out; __builtin_memcpy(&out, t, 32);
  return out;
}

// B fragment 32x16 bf16 from pair-interleaved tile: B[k][n] stored at
// base[(k>>1)*(2*ncols) + 2*n + (k&1)].  Order used here: N = lane&15; K = (lane>=16?16:0) + 2j + h.
static __device__ __forceinline__ v16bf load_b_pair_frag(const unsigned short* base, int ncols,
                                                         int nbase, int lane) {
  unsigned tmp32[8];
  int n = nbase + (lane & 15);
  int kp0 = (lane & 16) ? 8 : 0;   // (K base)/2
#pragma unroll
  for (int j = 0; j < 8; ++j) {
    tmp32[j] = *(const unsigned*)&base[(kp0 + j) * (2 * ncols) + 2 * n]; // (k,n),(k+1,n)
  }
  v16bf out; __builtin_memcpy(&out, tmp32, 32);
  return out;
}

// B fragment where source rows are N (B = srcT): B[k][n] = base[(nbase+n)*stride + kbase+k].
// Per lane: one contiguous 32-byte run -> 2x ds_load_b128.
static __device__ __forceinline__ v16bf load_bT_frag(const unsigned short* base, int stride,
                                                     int kbase, int nbase, int lane) {
  int n = nbase + (lane & 15);
  int ko = (lane & 16) ? 16 : 0;
  const unsigned short* p = &base[n * stride + kbase + ko];
  uint4 t[2];
  t[0] = *(const uint4*)p;
  t[1] = *(const uint4*)(p + 8);
  v16bf out; __builtin_memcpy(&out, t, 32);
  return out;
}

// ---------------- K0: fp32 -> bf16 conversions ----------------
__global__ void k_convert(const float* __restrict__ x, const float* __restrict__ wqk,
                          const float* __restrict__ wv, const float* __restrict__ wo,
                          unsigned short* __restrict__ xb, unsigned short* __restrict__ wqkb,
                          unsigned short* __restrict__ wvb, unsigned short* __restrict__ wob) {
  int i = blockIdx.x * blockDim.x + threadIdx.x;
  const int n = BT * Cc;
  for (; i < n; i += gridDim.x * blockDim.x) {
    xb[i] = f2bf_u(x[i]);
    if (i < Cc * Cc) { wqkb[i] = f2bf_u(wqk[i]); wvb[i] = f2bf_u(wv[i]); wob[i] = f2bf_u(wo[i]); }
  }
}

// ---------------- K1: qk/v projection GEMM (WMMA bf16) ----------------
// out columns [0,512) -> qk via Wqk ; [512,1024) -> v via Wv ; head-major store (bh,t,dh)
__global__ void __launch_bounds__(128) k_proj_gemm(const unsigned short* __restrict__ xb,
    const unsigned short* __restrict__ wqkb, const unsigned short* __restrict__ wvb,
    unsigned short* __restrict__ qkb, unsigned short* __restrict__ vb) {
  __shared__ __align__(16) unsigned short At[64 * 32];
  __shared__ __align__(16) unsigned short Bt[32 * 64];  // pair-interleaved [k/2][n][k&1]
  int tid = threadIdx.x, lane = tid & 31, wid = tid >> 5;
  int m0 = blockIdx.x * 64;
  int g0 = blockIdx.y * 64;
  int which = g0 >> 9;
  int n0 = g0 & 511;
  const unsigned short* W = which ? wvb : wqkb;
  v8f acc[4];
#pragma unroll
  for (int nt = 0; nt < 4; ++nt) acc[nt] = v8f_zero();
  for (int kk = 0; kk < Cc; kk += 32) {
    if (kk + 32 < Cc)
      __builtin_prefetch(&xb[(size_t)(m0 + (tid >> 1)) * Cc + kk + 32], 0, 3);
    for (int idx = tid; idx < 64 * 32; idx += 128) {
      int m = idx >> 5, k = idx & 31;
      At[idx] = xb[(m0 + m) * Cc + kk + k];
    }
    for (int idx = tid; idx < 32 * 64; idx += 128) {
      int k = idx >> 6, nn = idx & 63;
      Bt[(k >> 1) * 128 + nn * 2 + (k & 1)] = W[(kk + k) * Cc + n0 + nn];
    }
    __syncthreads();
    v16bf a = load_a_frag(At, 32, wid * 16, 0, lane);
#pragma unroll
    for (int nt = 0; nt < 4; ++nt) {
      v16bf b = load_b_pair_frag(Bt, 64, nt * 16, lane);
      acc[nt] = wmma_bf16(a, b, acc[nt]);
    }
    __syncthreads();
  }
  int hg = (lane >> 4) & 1;
#pragma unroll
  for (int nt = 0; nt < 4; ++nt) {
    int f = n0 + nt * 16 + (lane & 15);
    int h = f >> 6, d = f & 63;
#pragma unroll
    for (int r = 0; r < 8; ++r) {
      int m = m0 + wid * 16 + r + 8 * hg;
      int b = m >> 12, t = m & (Tt - 1);
      unsigned short val = f2bf_u(acc[nt][r]);
      int dst = ((b * 8 + h) * Tt + t) * DH + d;
      if (which) vb[dst] = val; else qkb[dst] = val;
    }
  }
}

// ---------------- K2: LSH bucketing + stable counting sort (one WGP per bh) ----------------
__global__ void __launch_bounds__(256) k_hash_sort(const unsigned short* __restrict__ qkb,
    const float* __restrict__ rot, unsigned* __restrict__ sorted_flat) {
  __shared__ __align__(16) float rotL[DH * NH * (NB / 2)];   // 8192 floats
  __shared__ __align__(16) unsigned char bktL[FLATN];        // 16 KB
  __shared__ unsigned cnt[256];
  __shared__ unsigned start[256];
  int bh = blockIdx.x, tid = threadIdx.x;
  for (int i = tid; i < DH * NH * (NB / 2); i += 256) rotL[i] = rot[i];
  cnt[tid] = 0;
  __syncthreads();
  for (int t = tid; t < Tt; t += 256) {
    float q[DH];
    const unsigned short* row = &qkb[(bh * Tt + t) * DH];
    for (int d = 0; d < DH; ++d) q[d] = bf2f_u(row[d]);
    for (int h = 0; h < NH; ++h) {
      float best = -3.4e38f; int bi = 0;
      for (int i = 0; i < NB / 2; ++i) {
        float r = 0.f;
        for (int d = 0; d < DH; ++d) r += q[d] * rotL[(d * NH + h) * (NB / 2) + i];
        if (r > best)  { best = r;  bi = i; }
        if (-r > best) { best = -r; bi = (NB / 2) + i; }
      }
      bktL[h * Tt + t] = (unsigned char)(bi + h * NB);
    }
  }
  __syncthreads();
  for (int i = tid; i < FLATN; i += 256) atomicAdd(&cnt[bktL[i]], 1u);
  __syncthreads();
  if (tid == 0) {
    unsigned s = 0;
    for (int b = 0; b < 256; ++b) { start[b] = s; s += cnt[b]; }
  }
  __syncthreads();
  {
    unsigned pos = start[tid];
    unsigned char me = (unsigned char)tid;
    unsigned* out = &sorted_flat[bh * FLATN];
    for (int i = 0; i < FLATN; ++i)
      if (bktL[i] == me) out[pos++] = (unsigned)i;   // stable: increasing i within bucket
  }
}

// ---------------- K3: chunked attention with look-one-back (WMMA) ----------------
__global__ void __launch_bounds__(128) k_attn(const unsigned short* __restrict__ qkb,
    const unsigned short* __restrict__ vbuf, const unsigned* __restrict__ sorted_flat,
    float* __restrict__ logits, float* __restrict__ ohash) {
  __shared__ __align__(16) unsigned short qrow[64 * DH];
  __shared__ __align__(16) unsigned short krow[128 * DH];
  __shared__ __align__(16) unsigned short vrowT[DH * 128];  // transposed: [d][token]
  __shared__ __align__(16) unsigned short pmat[64 * 128];
  __shared__ int qflat[64];
  __shared__ int qpos[64];
  __shared__ int kpos[128];
  int chunk = blockIdx.x, bh = blockIdx.y;
  int tid = threadIdx.x, lane = tid & 31, wid = tid >> 5;
  const unsigned* sf = &sorted_flat[bh * FLATN];
  int prev = (chunk + NCHUNK - 1) % NCHUNK;
  if (tid < 64) {
    int i = (int)sf[chunk * 64 + tid];
    qflat[tid] = i; qpos[tid] = i & (Tt - 1);
  }
  if (tid < 128) {
    int p = (tid < 64) ? (prev * 64 + tid) : (chunk * 64 + (tid - 64));
    kpos[tid] = ((int)sf[p]) & (Tt - 1);
  }
  __syncthreads();
  for (int idx = tid; idx < 64 * DH; idx += 128) {
    int j = idx >> 6, d = idx & 63;
    qrow[idx] = qkb[(bh * Tt + qpos[j]) * DH + d];
  }
  for (int idx = tid; idx < 128 * DH; idx += 128) {
    int j = idx >> 6, d = idx & 63;
    int t = kpos[j];
    krow[idx] = qkb[(bh * Tt + t) * DH + d];
    vrowT[d * 128 + j] = vbuf[(bh * Tt + t) * DH + d];   // transpose on the fly
  }
  __syncthreads();
  { // L2-normalize key rows (one row per thread)
    int j = tid;
    float ss = 0.f;
    for (int d = 0; d < DH; ++d) { float x = bf2f_u(krow[j * DH + d]); ss += x * x; }
    float sc = 1.f / fmaxf(sqrtf(ss), 1e-12f);
    for (int d = 0; d < DH; ++d) krow[j * DH + d] = f2bf_u(bf2f_u(krow[j * DH + d]) * sc);
  }
  __syncthreads();
  int mbase = wid * 16;
  v8f accD[8];
#pragma unroll
  for (int nt = 0; nt < 8; ++nt) accD[nt] = v8f_zero();
  for (int kk = 0; kk < DH; kk += 32) {
    v16bf a = load_a_frag(qrow, DH, mbase, kk, lane);
#pragma unroll
    for (int nt = 0; nt < 8; ++nt) {
      v16bf b = load_bT_frag(krow, DH, kk, nt * 16, lane);   // B = K^T, b128 loads
      accD[nt] = wmma_bf16(a, b, accD[nt]);
    }
  }
  int hg = (lane >> 4) & 1;
#pragma unroll
  for (int nt = 0; nt < 8; ++nt) {
    int ct = nt * 16 + (lane & 15);
    int kp = kpos[ct];
#pragma unroll
    for (int r = 0; r < 8; ++r) {
      int row = mbase + r + 8 * hg;
      float v = accD[nt][r] * 0.125f;        // dim^-0.5, dim=64
      if (qpos[row] == kp) v = -5e4f;        // TOKEN_SELF_ATTN_VALUE
      accD[nt][r] = v;
    }
  }
  float lse_r[8];
#pragma unroll
  for (int r = 0; r < 8; ++r) {
    float mx = -3.4e38f;
#pragma unroll
    for (int nt = 0; nt < 8; ++nt) mx = fmaxf(mx, accD[nt][r]);
    for (int m = 1; m < 16; m <<= 1) mx = fmaxf(mx, __shfl_xor(mx, m, 32));
    float s = 0.f;
#pragma unroll
    for (int nt = 0; nt < 8; ++nt) s += __expf(accD[nt][r] - mx);
    for (int m = 1; m < 16; m <<= 1) s += __shfl_xor(s, m, 32);
    float inv = 1.f / s;
    int row = mbase + r + 8 * hg;
#pragma unroll
    for (int nt = 0; nt < 8; ++nt) {
      float p = __expf(accD[nt][r] - mx) * inv;
      pmat[row * 128 + nt * 16 + (lane & 15)] = f2bf_u(p);
    }
    lse_r[r] = mx + __logf(s);
  }
  if ((lane & 15) == 0) {
#pragma unroll
    for (int r = 0; r < 8; ++r) {
      int row = mbase + r + 8 * hg;
      logits[bh * FLATN + qflat[row]] = lse_r[r];
    }
  }
  __syncthreads();
  v8f accO[4];
#pragma unroll
  for (int nt = 0; nt < 4; ++nt) accO[nt] = v8f_zero();
  for (int kk = 0; kk < 128; kk += 32) {
    v16bf a = load_a_frag(pmat, 128, mbase, kk, lane);
#pragma unroll
    for (int nt = 0; nt < 4; ++nt) {
      v16bf b = load_bT_frag(vrowT, 128, kk, nt * 16, lane);  // B[k=tok][n=d] = vrowT[d][tok]
      accO[nt] = wmma_bf16(a, b, accO[nt]);
    }
  }
#pragma unroll
  for (int nt = 0; nt < 4; ++nt) {
    int d = nt * 16 + (lane & 15);
#pragma unroll
    for (int r = 0; r < 8; ++r) {
      int row = mbase + r + 8 * hg;
      ohash[(size_t)(bh * FLATN + qflat[row]) * DH + d] = accO[nt][r];  // unsort scatter
    }
  }
}

// ---------------- K4: combine hash rounds, weighted by logsumexp ----------------
__global__ void __launch_bounds__(256) k_combine(const float* __restrict__ logits,
    const float* __restrict__ ohash, unsigned short* __restrict__ ocomb) {
  int tid = threadIdx.x, lane = tid & 31;
  int row = blockIdx.x * 8 + (tid >> 5);  // (bh, t) flattened, 32*4096 rows
  int bh = row >> 12, t = row & (Tt - 1);
  float l[NH], mx = -3.4e38f;
  for (int h = 0; h < NH; ++h) { l[h] = logits[bh * FLATN + h * Tt + t]; mx = fmaxf(mx, l[h]); }
  float s = 0.f, w[NH];
  for (int h = 0; h < NH; ++h) { w[h] = __expf(l[h] - mx); s += w[h]; }
  float inv = 1.f / s;
  int b = bh >> 3, head = bh & 7;
  for (int dd = lane * 2; dd < lane * 2 + 2; ++dd) {
    float acc = 0.f;
    for (int h = 0; h < NH; ++h)
      acc += w[h] * ohash[(size_t)(bh * FLATN + h * Tt + t) * DH + dd];
    ocomb[(b * Tt + t) * Cc + head * DH + dd] = f2bf_u(acc * inv);
  }
}

// ---------------- K5: output projection + bias (WMMA), f32 out ----------------
__global__ void __launch_bounds__(128) k_final_gemm(const unsigned short* __restrict__ ob,
    const unsigned short* __restrict__ wob, const float* __restrict__ bias,
    float* __restrict__ out) {
  __shared__ __align__(16) unsigned short At[64 * 32];
  __shared__ __align__(16) unsigned short Bt[32 * 64];  // pair-interleaved
  int tid = threadIdx.x, lane = tid & 31, wid = tid >> 5;
  int m0 = blockIdx.x * 64, n0 = blockIdx.y * 64;
  v8f acc[4];
#pragma unroll
  for (int nt = 0; nt < 4; ++nt) acc[nt] = v8f_zero();
  for (int kk = 0; kk < Cc; kk += 32) {
    if (kk + 32 < Cc)
      __builtin_prefetch(&ob[(size_t)(m0 + (tid >> 1)) * Cc + kk + 32], 0, 3);
    for (int idx = tid; idx < 64 * 32; idx += 128) {
      int m = idx >> 5, k = idx & 31;
      At[idx] = ob[(m0 + m) * Cc + kk + k];
    }
    for (int idx = tid; idx < 32 * 64; idx += 128) {
      int k = idx >> 6, nn = idx & 63;
      Bt[(k >> 1) * 128 + nn * 2 + (k & 1)] = wob[(kk + k) * Cc + n0 + nn];
    }
    __syncthreads();
    v16bf a = load_a_frag(At, 32, wid * 16, 0, lane);
#pragma unroll
    for (int nt = 0; nt < 4; ++nt) {
      v16bf b = load_b_pair_frag(Bt, 64, nt * 16, lane);
      acc[nt] = wmma_bf16(a, b, acc[nt]);
    }
    __syncthreads();
  }
  int hg = (lane >> 4) & 1;
#pragma unroll
  for (int nt = 0; nt < 4; ++nt) {
    int c = n0 + nt * 16 + (lane & 15);
    float bb = bias[c];
#pragma unroll
    for (int r = 0; r < 8; ++r) {
      int m = m0 + wid * 16 + r + 8 * hg;
      out[(size_t)m * Cc + c] = acc[nt][r] + bb;
    }
  }
}

extern "C" void kernel_launch(void* const* d_in, const int* in_sizes, int n_in,
                              void* d_out, int out_size, void* d_ws, size_t ws_size,
                              hipStream_t stream) {
  (void)in_sizes; (void)n_in; (void)out_size; (void)ws_size;
  const float* queries = (const float*)d_in[0];
  const float* Wqk = (const float*)d_in[6];
  const float* Wv  = (const float*)d_in[7];
  const float* Wo  = (const float*)d_in[8];
  const float* bo  = (const float*)d_in[9];
  const float* rot = (const float*)d_in[10];

  char* ws = (char*)d_ws;
  size_t off = 0;
  unsigned short* xb   = (unsigned short*)(ws + off); off += (size_t)BT * Cc * 2;
  unsigned short* wqkb = (unsigned short*)(ws + off); off += (size_t)Cc * Cc * 2;
  unsigned short* wvb  = (unsigned short*)(ws + off); off += (size_t)Cc * Cc * 2;
  unsigned short* wob  = (unsigned short*)(ws + off); off += (size_t)Cc * Cc * 2;
  unsigned short* qkb  = (unsigned short*)(ws + off); off += (size_t)BH * Tt * DH * 2;
  unsigned short* vb   = (unsigned short*)(ws + off); off += (size_t)BH * Tt * DH * 2;
  unsigned*       sf   = (unsigned*)(ws + off);       off += (size_t)BH * FLATN * 4;
  float*          lg   = (float*)(ws + off);          off += (size_t)BH * FLATN * 4;
  float*          oh   = (float*)(ws + off);          off += (size_t)BH * FLATN * DH * 4;
  unsigned short* oc   = (unsigned short*)(ws + off); off += (size_t)BT * Cc * 2;

  k_convert<<<(BT * Cc + 255) / 256, 256, 0, stream>>>(queries, Wqk, Wv, Wo, xb, wqkb, wvb, wob);
  k_proj_gemm<<<dim3(BT / 64, 1024 / 64), 128, 0, stream>>>(xb, wqkb, wvb, qkb, vb);
  k_hash_sort<<<BH, 256, 0, stream>>>(qkb, rot, sf);
  k_attn<<<dim3(NCHUNK, BH), 128, 0, stream>>>(qkb, vb, sf, lg, oh);
  k_combine<<<(BH * Tt) / 8, 256, 0, stream>>>(lg, oh, oc);
  k_final_gemm<<<dim3(BT / 64, Cc / 64), 128, 0, stream>>>(oc, wob, bo, (float*)d_out);
}
